// MultiHeadAttention_1314259993118
// MI455X (gfx1250) — matrix operand order repaired and checked by reading, not compiled, not run on hardware
//
#include <hip/hip_runtime.h>
#include <hip/hip_bf16.h>
#include <stdint.h>

// Problem sizes (match the reference)
#define Bq 4
#define Sq 1024
#define Eq 1024
#define Hq 16
#define Dq 64
#define Mq (Bq * Sq)   // 4096 rows of x

typedef __attribute__((ext_vector_type(16))) __bf16        v16bf;
typedef __attribute__((ext_vector_type(8)))  float         v8f;
typedef __attribute__((ext_vector_type(8)))  unsigned int  v8u;

static __device__ __forceinline__ unsigned short f2bf(float f) {
  unsigned int u = __builtin_bit_cast(unsigned int, f);
  unsigned int r = u + 0x7FFFu + ((u >> 16) & 1u);   // round-to-nearest-even
  return (unsigned short)(r >> 16);
}
static __device__ __forceinline__ unsigned int pack2(unsigned short lo, unsigned short hi) {
  return (unsigned int)lo | ((unsigned int)hi << 16);
}
static __device__ __forceinline__ v8f vzero() {
  v8f z;
#pragma unroll
  for (int i = 0; i < 8; ++i) z[i] = 0.0f;
  return z;
}

// ---- WMMA fragment loaders (CDNA5 ISA 7.12.2 layouts, wave32) ----

// A-matrix 16x32 bf16: lane r=lane&15 selects row, half=lane>>4.
// dword v holds K pair at col ((v/4)*16 + half*8 + (v%4)*2). Pairs contiguous -> b32 loads.
static __device__ __forceinline__ v8u load_a_frag(const unsigned short* __restrict__ p, int ld) {
  int lane = threadIdx.x & 31;
  const unsigned short* row = p + (size_t)(lane & 15) * ld;
  int half = lane >> 4;
  v8u a;
#pragma unroll
  for (int v = 0; v < 8; ++v) {
    int c = ((v >> 2) << 4) + (half << 3) + ((v & 3) << 1);
    a[v] = *(const unsigned int*)(row + c);
  }
  return a;
}

// B-matrix 32x16 bf16 read from "Bt" storage: B[k][n] = p[n*ld + k].
// lane n=lane&15 selects column, dword v holds K pair ((v/4)*16 + half*8 + (v%4)*2) -> contiguous b32 loads.
static __device__ __forceinline__ v8u load_b_frag(const unsigned short* __restrict__ p, int ld) {
  int lane = threadIdx.x & 31;
  const unsigned short* row = p + (size_t)(lane & 15) * ld;
  int half = lane >> 4;
  v8u b;
#pragma unroll
  for (int v = 0; v < 8; ++v) {
    int k = ((v >> 2) << 4) + (half << 3) + ((v & 3) << 1);
    b[v] = *(const unsigned int*)(row + k);
  }
  return b;
}

// A-matrix 16x32 built from f32 probabilities in LDS (row stride Sq), converted to bf16.
static __device__ __forceinline__ v8u load_a_frag_lds(const float* __restrict__ p) {
  int lane = threadIdx.x & 31;
  const float* row = p + (lane & 15) * Sq;
  int half = lane >> 4;
  v8u a;
#pragma unroll
  for (int v = 0; v < 8; ++v) {
    int c = ((v >> 2) << 4) + (half << 3) + ((v & 3) << 1);
    a[v] = pack2(f2bf(row[c]), f2bf(row[c + 1]));
  }
  return a;
}

static __device__ __forceinline__ v8f wmma_bf16(v8u a, v8u b, v8f c) {
  return __builtin_amdgcn_wmma_f32_16x16x32_bf16(
      false, __builtin_bit_cast(v16bf, a),
      false, __builtin_bit_cast(v16bf, b),
      (short)0, c, false, false);
}

// ---- elementwise fp32 -> bf16 ----
__global__ void cvt_f32_bf16(const float* __restrict__ in, unsigned short* __restrict__ out, int n) {
  int i = blockIdx.x * blockDim.x + threadIdx.x;
  if (i < n) out[i] = f2bf(in[i]);
}

// ---- fp32 [rows x cols] -> bf16 transposed [cols x rows] (LDS-tiled) ----
__global__ __launch_bounds__(1024) void transpose_f32_bf16(
    const float* __restrict__ in, unsigned short* __restrict__ out, int rows, int cols) {
  __shared__ float tile[32][33];
  int tx = threadIdx.x, ty = threadIdx.y;
  int c0 = blockIdx.x << 5, r0 = blockIdx.y << 5;
  tile[ty][tx] = in[(size_t)(r0 + ty) * cols + (c0 + tx)];
  __syncthreads();
  out[(size_t)(c0 + ty) * rows + (r0 + tx)] = f2bf(tile[tx][ty]);
}

// ---- 64x64-tile WMMA GEMM: C = alpha * (A1@B1t^T [+ A2@B2t^T]) ----
// A: [M x K] bf16 row-major; Bt: [N x K] bf16 row-major (i.e. B transposed).
// STORE_MODE: 0 = f32 row-major [M x N]
//             1 = bf16 row-major [M x N]
//             2 = bf16 per-head transposed: out[((b*Hq + h)*Dq + d)*Sq + s]
//                 with m = b*Sq + s, c = h*Dq + d   (for V tensors consumed as WMMA B)
template <int STORE_MODE, bool DUAL>
__global__ __launch_bounds__(32) void gemm64x64(
    const unsigned short* __restrict__ A1, const unsigned short* __restrict__ B1t,
    const unsigned short* __restrict__ A2, const unsigned short* __restrict__ B2t,
    void* __restrict__ Cout, int M, int N, int K, float alpha) {
  int lane = threadIdx.x & 31;
  int n0 = blockIdx.x << 6, m0 = blockIdx.y << 6;
  v8f acc[4][4];
#pragma unroll
  for (int i = 0; i < 4; ++i)
#pragma unroll
    for (int j = 0; j < 4; ++j) acc[i][j] = vzero();

  for (int k0 = 0; k0 < K; k0 += 32) {
    if (k0 + 128 < K) {  // stream-ahead hints (global_prefetch_b8)
      __builtin_prefetch(A1 + (size_t)(m0 + (lane & 3) * 16 + (lane >> 2)) * K + k0 + 128, 0, 0);
      __builtin_prefetch(B1t + (size_t)(n0 + (lane & 3) * 16 + (lane >> 2)) * K + k0 + 128, 0, 0);
    }
    v8u af[4], bf[4];
#pragma unroll
    for (int i = 0; i < 4; ++i) af[i] = load_a_frag(A1 + (size_t)(m0 + i * 16) * K + k0, K);
#pragma unroll
    for (int j = 0; j < 4; ++j) bf[j] = load_b_frag(B1t + (size_t)(n0 + j * 16) * K + k0, K);
#pragma unroll
    for (int i = 0; i < 4; ++i)
#pragma unroll
      for (int j = 0; j < 4; ++j) acc[i][j] = wmma_bf16(af[i], bf[j], acc[i][j]);
    if (DUAL) {
#pragma unroll
      for (int i = 0; i < 4; ++i) af[i] = load_a_frag(A2 + (size_t)(m0 + i * 16) * K + k0, K);
#pragma unroll
      for (int j = 0; j < 4; ++j) bf[j] = load_b_frag(B2t + (size_t)(n0 + j * 16) * K + k0, K);
#pragma unroll
      for (int i = 0; i < 4; ++i)
#pragma unroll
        for (int j = 0; j < 4; ++j) acc[i][j] = wmma_bf16(af[i], bf[j], acc[i][j]);
    }
  }
  int n = lane & 15, half = lane >> 4;
#pragma unroll
  for (int i = 0; i < 4; ++i)
#pragma unroll
    for (int j = 0; j < 4; ++j)
#pragma unroll
      for (int v = 0; v < 8; ++v) {
        int m = m0 + i * 16 + v + (half << 3);   // global row  (b*Sq + s)
        int c = n0 + j * 16 + n;                 // global col  (h*Dq + d)
        float val = acc[i][j][v] * alpha;
        if (STORE_MODE == 0) {
          ((float*)Cout)[(size_t)m * N + c] = val;
        } else if (STORE_MODE == 1) {
          ((unsigned short*)Cout)[(size_t)m * N + c] = f2bf(val);
        } else {
          int b = m >> 10, s = m & (Sq - 1);
          size_t idx = ((size_t)c * Sq) * 0;  // (folded below; keep index math simple)
          idx = ((size_t)((b * Hq) << 6) + (size_t)c) * Sq + s;  // ((b*Hq+h)*Dq + d)*Sq + s
          ((unsigned short*)Cout)[idx] = f2bf(val);
        }
      }
}

// ---- fused dual-branch causal attention ----
// One wave per (b, h, 16-row query tile). Full 16x1024 logit stripe in LDS (64 KB).
// vb / v2b are in per-head transposed layout: vt[((b*Hq+h)*Dq + d)*Sq + t].
__global__ __launch_bounds__(32) void attn_fused(
    const unsigned short* __restrict__ qb,  const unsigned short* __restrict__ kb,
    const unsigned short* __restrict__ vb,  const unsigned short* __restrict__ q2b,
    const unsigned short* __restrict__ k2b, const unsigned short* __restrict__ v2b,
    float* __restrict__ attn, unsigned short* __restrict__ r1, unsigned short* __restrict__ r2) {
  __shared__ float P[16 * Sq];   // 64 KB probability row-stripe (WGP has 320 KB)
  int lane = threadIdx.x & 31;
  int qt = blockIdx.x, h = blockIdx.y, b = blockIdx.z;
  int q0 = qt << 4;
  size_t rowb = (size_t)b * Sq;
  int hoff = h * Dq;
  int n = lane & 15, half = lane >> 4;

  // Q fragments for both branches (D=64 -> two K=32 chunks each)
  v8u aq[2], aq2[2];
#pragma unroll
  for (int c = 0; c < 2; ++c) {
    aq[c]  = load_a_frag(qb  + (rowb + q0) * Eq + hoff + c * 32, Eq);
    aq2[c] = load_a_frag(q2b + (rowb + q0) * Eq + hoff + c * 32, Eq);
  }

  int ncols = (qt + 1) << 4;   // causal: valid key columns [0, ncols)

  // logits = 0.5*(q k^T + q2 k2^T)/sqrt(D); scale = 0.5/8 = 0.0625
  for (int kt = 0; kt <= qt; ++kt) {
    int t0 = kt << 4;
    const unsigned short* K1 = kb  + (rowb + t0) * Eq + hoff;
    const unsigned short* K2 = k2b + (rowb + t0) * Eq + hoff;
    v8f s = vzero();
#pragma unroll
    for (int c = 0; c < 2; ++c) {
      s = wmma_bf16(aq[c],  load_b_frag(K1 + c * 32, Eq), s);
      s = wmma_bf16(aq2[c], load_b_frag(K2 + c * 32, Eq), s);
    }
#pragma unroll
    for (int v = 0; v < 8; ++v) {
      int m = v + (half << 3);
      int t = t0 + n;
      P[m * Sq + t] = (t <= q0 + m) ? s[v] * 0.0625f : -1e30f;
    }
  }
  // zero-pad up to a 32-wide chunk boundary for the P@V stage
  if (ncols & 31) {
    for (int i = lane; i < 256; i += 32) P[(i >> 4) * Sq + ncols + (i & 15)] = 0.0f;
  }
  __syncthreads();  // single-wave workgroup: lowers to S_NOP, keeps LDS ordering explicit

  // exact softmax per row (32 lanes cooperate), write attn probabilities once
  float* attnBase = attn + (((size_t)(b * Hq + h) * Sq + q0) * Sq);
  for (int r = 0; r < 16; ++r) {
    float* row = P + r * Sq;
    float mx = -1e30f;
    for (int c = lane; c < ncols; c += 32) mx = fmaxf(mx, row[c]);
#pragma unroll
    for (int o = 16; o; o >>= 1) mx = fmaxf(mx, __shfl_xor(mx, o, 32));
    float sum = 0.0f;
    for (int c = lane; c < ncols; c += 32) {
      float e = __expf(row[c] - mx);
      row[c] = e;
      sum += e;
    }
#pragma unroll
    for (int o = 16; o; o >>= 1) sum += __shfl_xor(sum, o, 32);
    float inv = 1.0f / sum;
    float* arow = attnBase + (size_t)r * Sq;
    for (int c = lane; c < ncols; c += 32) {
      float p = row[c] * inv;
      row[c] = p;
      arow[c] = p;
    }
    for (int c = ncols + lane; c < Sq; c += 32) arow[c] = 0.0f;  // strict upper triangle
  }
  __syncthreads();

  // O = P@V, O2 = P@V2. V in vt layout -> same packed-pair B loads as the K path.
  const unsigned short* Vt1 = vb  + ((size_t)(b * Hq + h) * Dq) * Sq;
  const unsigned short* Vt2 = v2b + ((size_t)(b * Hq + h) * Dq) * Sq;
  v8f acc[4], acc2[4];
#pragma unroll
  for (int j = 0; j < 4; ++j) { acc[j] = vzero(); acc2[j] = vzero(); }
  int nch = (ncols + 31) >> 5;
  for (int ch = 0; ch < nch; ++ch) {
    int t0 = ch << 5;
    v8u aP = load_a_frag_lds(P + t0);
#pragma unroll
    for (int j = 0; j < 4; ++j) {
      acc[j]  = wmma_bf16(aP, load_b_frag(Vt1 + (size_t)(j * 16) * Sq + t0, Sq), acc[j]);
      acc2[j] = wmma_bf16(aP, load_b_frag(Vt2 + (size_t)(j * 16) * Sq + t0, Sq), acc2[j]);
    }
  }
  // store per-head results as bf16 rows of [B*S, H*D]
#pragma unroll
  for (int j = 0; j < 4; ++j)
#pragma unroll
    for (int v = 0; v < 8; ++v) {
      int m = v + (half << 3);
      size_t idx = (rowb + q0 + m) * Eq + hoff + j * 16 + n;
      r1[idx] = f2bf(acc[j][v]);
      r2[idx] = f2bf(acc2[j][v]);
    }
}

extern "C" void kernel_launch(void* const* d_in, const int* in_sizes, int n_in,
                              void* d_out, int out_size, void* d_ws, size_t ws_size,
                              hipStream_t stream) {
  (void)in_sizes; (void)n_in; (void)out_size; (void)ws_size;
  const float* x = (const float*)d_in[0];
  const float* Wf[8];
  for (int i = 0; i < 8; ++i) Wf[i] = (const float*)d_in[1 + i];  // Wq,Wk,Wv,Wo,Wq_fm,Wk_fm,Wv_fm,Wo_fm

  // workspace carve-out (bf16 elements): ~88 MB total
  unsigned short* ws = (unsigned short*)d_ws;
  size_t off = 0;
  unsigned short* xbf = ws + off; off += (size_t)Mq * Eq;
  unsigned short* wt[8];
  for (int i = 0; i < 8; ++i) { wt[i] = ws + off; off += (size_t)Eq * Eq; }
  unsigned short* qkv[6];   // 0:q 1:k 2:v(vt layout) 3:q2 4:k2 5:v2(vt layout)
  for (int i = 0; i < 6; ++i) { qkv[i] = ws + off; off += (size_t)Mq * Eq; }
  unsigned short* r1 = ws + off; off += (size_t)Mq * Eq;
  unsigned short* r2 = ws + off; off += (size_t)Mq * Eq;

  // 1) x -> bf16
  {
    int ntot = Mq * Eq;
    cvt_f32_bf16<<<(ntot + 255) / 256, 256, 0, stream>>>(x, xbf, ntot);
  }
  // 2) weights -> bf16 transposed [N x K]
  for (int i = 0; i < 8; ++i)
    transpose_f32_bf16<<<dim3(Eq / 32, Eq / 32), dim3(32, 32), 0, stream>>>(Wf[i], wt[i], Eq, Eq);

  // 3) projections: q,k,q2,k2 row-major bf16; v,v2 in per-head transposed layout
  dim3 ggrid(Eq / 64, Mq / 64);
  gemm64x64<1, false><<<ggrid, 32, 0, stream>>>(xbf, wt[0], nullptr, nullptr, qkv[0], Mq, Eq, Eq, 1.0f); // q
  gemm64x64<1, false><<<ggrid, 32, 0, stream>>>(xbf, wt[1], nullptr, nullptr, qkv[1], Mq, Eq, Eq, 1.0f); // k
  gemm64x64<2, false><<<ggrid, 32, 0, stream>>>(xbf, wt[2], nullptr, nullptr, qkv[2], Mq, Eq, Eq, 1.0f); // v  (vt)
  gemm64x64<1, false><<<ggrid, 32, 0, stream>>>(xbf, wt[4], nullptr, nullptr, qkv[3], Mq, Eq, Eq, 1.0f); // q2
  gemm64x64<1, false><<<ggrid, 32, 0, stream>>>(xbf, wt[5], nullptr, nullptr, qkv[4], Mq, Eq, Eq, 1.0f); // k2
  gemm64x64<2, false><<<ggrid, 32, 0, stream>>>(xbf, wt[6], nullptr, nullptr, qkv[5], Mq, Eq, Eq, 1.0f); // v2 (vt)

  // 4) fused dual-branch causal attention; attn written to second output region
  float* attn = (float*)d_out + (size_t)Mq * Eq;
  attn_fused<<<dim3(Sq / 16, Hq, Bq), 32, 0, stream>>>(
      qkv[0], qkv[1], qkv[2], qkv[3], qkv[4], qkv[5], attn, r1, r2);

  // 5) out = 0.5 * (R1 @ Wo + R2 @ Wo_fm), fp32 to d_out
  gemm64x64<0, true><<<ggrid, 32, 0, stream>>>(
      r1, wt[3], r2, wt[7], d_out, Mq, Eq, Eq, 0.5f);
}
